// CTLSTM_25537875542572
// MI455X (gfx1250) — matrix operand order repaired and checked by reading, not compiled, not run on hardware
//
#include <hip/hip_runtime.h>
#include <hip/hip_bf16.h>

// ---------------- problem dims ----------------
#define KDIM   128
#define HDIM   512
#define BDIM   64
#define NEV    512
#define NGATES 7
#define GCOLS  (NGATES * HDIM)   // 3584
#define NWG    32                // persistent workgroups; each owns 16 hidden units
#define NTHR   128               // 4 waves of 32

typedef __attribute__((ext_vector_type(16))) _Float16 v16h;
typedef __attribute__((ext_vector_type(8)))  _Float16 v8h;
typedef __attribute__((ext_vector_type(8)))  float    v8f;

// ---------------- workspace layout (bytes, all 256B aligned) ----------------
static constexpr size_t OFF_CNT   = 0;                       // 1 x u32 barrier counter
static constexpr size_t OFF_MARKS = 256;                     // int[B*N]
static constexpr size_t SZ_MARKS  = (size_t)BDIM * NEV * 4;  // 131072
static constexpr size_t OFF_WVHI  = OFF_MARKS + SZ_MARKS;
static constexpr size_t SZ_WV     = (size_t)HDIM * GCOLS * 2;    // 3670016
static constexpr size_t OFF_WVLO  = OFF_WVHI + SZ_WV;
static constexpr size_t OFF_WLHI  = OFF_WVLO + SZ_WV;
static constexpr size_t SZ_WL     = (size_t)HDIM * KDIM * 2;     // 131072
static constexpr size_t OFF_WLLO  = OFF_WLHI + SZ_WL;
static constexpr size_t OFF_HTHI  = OFF_WLLO + SZ_WL;
static constexpr size_t SZ_HT     = (size_t)2 * BDIM * HDIM * 2; // double buffered
static constexpr size_t OFF_HTLO  = OFF_HTHI + SZ_HT;
// total ~8.1 MB

__device__ __forceinline__ float sigf(float x) { return 1.f / (1.f + expf(-x)); }

__device__ __forceinline__ v16h cat8(v8h a, v8h b) {
    return __builtin_shufflevector(a, b, 0,1,2,3,4,5,6,7,8,9,10,11,12,13,14,15);
}

// ============================================================================
// init: reset barrier counter, recover marks from one-hot rows, zero ht bufs
// ============================================================================
__global__ void ctlstm_init_kernel(const float* __restrict__ seq,
                                   int* __restrict__ marks,
                                   _Float16* __restrict__ hthi,
                                   _Float16* __restrict__ htlo,
                                   unsigned* __restrict__ counter) {
    int idx = blockIdx.x * blockDim.x + threadIdx.x;
    int stride = gridDim.x * blockDim.x;
    if (idx == 0) *counter = 0u;
    for (int i = idx; i < BDIM * NEV; i += stride) {
        const float* row = seq + (size_t)i * KDIM;
        int m = 0;
        for (int k = 0; k < KDIM; ++k) {
            if (row[k] > 0.5f) { m = k; break; }
        }
        marks[i] = m;
    }
    for (int i = idx; i < 2 * BDIM * HDIM; i += stride) {
        hthi[i] = (_Float16)0.f;
        htlo[i] = (_Float16)0.f;
    }
}

// ============================================================================
// pack W_V / W_lamb (fp32) -> split fp16 hi/lo in WMMA B-fragment lane layout:
//   halfIdx = ((nn*16 + kk)*32 + lane)*16 + h
//   col  = nn*16 + (lane&15)
//   krow = kk*32 + (h<8 ? 8*(lane>>4) + h : 16 + 8*(lane>>4) + h-8)  (B is 32x16 f16 per tile)
// ============================================================================
__global__ void ctlstm_pack_kernel(const float* __restrict__ WV,
                                   const float* __restrict__ WL,
                                   _Float16* __restrict__ WVhi,
                                   _Float16* __restrict__ WVlo,
                                   _Float16* __restrict__ WLhi,
                                   _Float16* __restrict__ WLlo) {
    int idx = blockIdx.x * blockDim.x + threadIdx.x;
    int stride = gridDim.x * blockDim.x;
    for (int i = idx; i < HDIM * GCOLS; i += stride) {
        int h = i & 15, lane = (i >> 4) & 31, t2 = i >> 9;
        int kk = t2 & 15, nn = t2 >> 4;
        int col  = nn * 16 + (lane & 15);
        int krow = kk * 32 + ((h < 8) ? 0 : 8) + ((lane < 16) ? 0 : 8) + h;
        float w = WV[(size_t)krow * GCOLS + col];
        _Float16 hi = (_Float16)w;
        WVhi[i] = hi;
        WVlo[i] = (_Float16)(w - (float)hi);
    }
    for (int i = idx; i < HDIM * KDIM; i += stride) {
        int h = i & 15, lane = (i >> 4) & 31, t2 = i >> 9;
        int kk = t2 & 15, nn = t2 >> 4;   // nn in [0,8)
        int col  = nn * 16 + (lane & 15);
        int krow = kk * 32 + ((h < 8) ? 0 : 8) + ((lane < 16) ? 0 : 8) + h;
        float w = WL[(size_t)krow * KDIM + col];
        _Float16 hi = (_Float16)w;
        WLhi[i] = hi;
        WLlo[i] = (_Float16)(w - (float)hi);
    }
}

// ============================================================================
// lambda tile: this WG computes lamb[16*r2..+16, 16*c2..+16] for step t.
// K=512 split over the 4 waves, partials reduced through LDS.
// ============================================================================
__device__ __forceinline__ void lamb_tile(
    int t, int r2, int c2, int wave, int lane, int tid,
    const _Float16* __restrict__ hthi, const _Float16* __restrict__ htlo,
    const _Float16* __restrict__ WLhi, const _Float16* __restrict__ WLlo,
    const float* __restrict__ bLam, const float* __restrict__ scale,
    float (*lpart)[8][32], float* __restrict__ out_lamb) {
    const int halfsel = (lane >> 4) & 1;
    const int l15     = lane & 15;
    const int arow    = 16 * r2 + l15;
    const _Float16* Ah = hthi + (size_t)arow * HDIM;
    const _Float16* Al = htlo + (size_t)arow * HDIM;

    v8f acc = {};
#pragma unroll
    for (int q = 0; q < 4; ++q) {
        const int kk = wave * 4 + q;
        const int kb = kk * 32 + halfsel * 8;
        v16h ahi = cat8(*(const v8h*)(Ah + kb), *(const v8h*)(Ah + kb + 16));
        v16h alo = cat8(*(const v8h*)(Al + kb), *(const v8h*)(Al + kb + 16));
        const size_t boff = ((size_t)(c2 * 16 + kk) * 32 + lane) * 16;
        v16h bhi = *(const v16h*)(WLhi + boff);
        v16h blo = *(const v16h*)(WLlo + boff);
        acc = __builtin_amdgcn_wmma_f32_16x16x32_f16(false, ahi, false, bhi, (short)0, acc, false, false);
        acc = __builtin_amdgcn_wmma_f32_16x16x32_f16(false, alo, false, bhi, (short)0, acc, false, false);
        acc = __builtin_amdgcn_wmma_f32_16x16x32_f16(false, ahi, false, blo, (short)0, acc, false, false);
    }
#pragma unroll
    for (int v = 0; v < 8; ++v) lpart[wave][v][lane] = acc[v];
    __syncthreads();

    for (int e = tid; e < 256; e += NTHR) {
        int v = e >> 5, l = e & 31;
        float s = lpart[0][v][l] + lpart[1][v][l] + lpart[2][v][l] + lpart[3][v][l];
        int m = v + ((l >= 16) ? 8 : 0);
        int n = l & 15;
        int b = 16 * r2 + m;
        int k = 16 * c2 + n;
        float til = s + bLam[k];
        float sc  = scale[k];
        float u   = til / sc;
        float sp  = (u > 0.f) ? (u + log1pf(expf(-u))) : log1pf(expf(u));
        out_lamb[((size_t)b * NEV + t) * KDIM + k] = sc * sp;
    }
    __syncthreads();
}

// ============================================================================
// persistent recurrent kernel: 32 WGs x 128 threads, grid barrier per step.
// The WG's W_V-hi slice (112 KB) lives in LDS for all 512 steps; W_V-lo is
// streamed from L2 so DS and VMEM pipes are both loaded.
// ============================================================================
__global__ __launch_bounds__(NTHR, 1)
void ctlstm_persistent_kernel(
    const float* __restrict__ times, const float* __restrict__ WU,
    const float* __restrict__ bU,    const float* __restrict__ bV,
    const float* __restrict__ bLam,  const float* __restrict__ scale,
    const int* __restrict__ marks,
    const _Float16* __restrict__ WVhi, const _Float16* __restrict__ WVlo,
    const _Float16* __restrict__ WLhi, const _Float16* __restrict__ WLlo,
    _Float16* __restrict__ HTHI, _Float16* __restrict__ HTLO,
    float* __restrict__ out_lamb,  float* __restrict__ out_clow,
    float* __restrict__ out_cbar,  float* __restrict__ out_delta,
    float* __restrict__ out_o,
    unsigned* __restrict__ counter) {
    // 114688 B: W_V-hi slice, fragment layout [(g*16+kk)*512 + lane*16 + h]
    __shared__ __align__(32) _Float16 wvh_s[NGATES * 16 * 512];
    __shared__ float gsm[BDIM][NGATES * 16];  // gate GEMM results for this slice
    __shared__ float ct_s[BDIM][16];          // owned c state
    __shared__ float cb_s[BDIM][16];          // owned cbar state
    __shared__ float lpart[4][8][32];         // lambda partials
    // total static LDS = 155,648 B

    const int w    = blockIdx.x;       // 0..31 -> hidden units [16w, 16w+16)
    const int tid  = threadIdx.x;
    const int wave = tid >> 5;         // row-tile (batch rows 16*wave..)
    const int lane = tid & 31;
    const int j0   = w * 16;
    const int r2   = w >> 3;           // lambda row-tile
    const int c2   = w & 7;            // lambda col-tile
    const int halfsel = (lane >> 4) & 1;
    const int l15     = lane & 15;

    // ---- one-time: stage this WG's W_V-hi slice into LDS (7 x 16 KB) ----
#pragma unroll
    for (int g = 0; g < NGATES; ++g) {
        const _Float16* src = WVhi + (size_t)(g * 32 + w) * 16 * 512;
        _Float16* dst = wvh_s + (size_t)g * 16 * 512;
        for (int i = tid; i < (16 * 512) / 8; i += NTHR)
            *(v8h*)(dst + i * 8) = *(const v8h*)(src + i * 8);
    }
    for (int e = tid; e < BDIM * 16; e += NTHR) {
        ct_s[e >> 4][e & 15] = 0.f;
        cb_s[e >> 4][e & 15] = 0.f;
    }
    __syncthreads();

    for (int t = 0; t < NEV; ++t) {
        const _Float16* hthi = HTHI + (size_t)(t & 1) * (BDIM * HDIM);
        const _Float16* htlo = HTLO + (size_t)(t & 1) * (BDIM * HDIM);

        // lambda for the previous step overlaps this step's gate GEMM
        if (t > 0)
            lamb_tile(t - 1, r2, c2, wave, lane, tid, hthi, htlo,
                      WLhi, WLlo, bLam, scale, lpart, out_lamb);

        // ---- gate GEMM: rows [16*wave,+16), cols {g*512 + j0 + n} ----
        v8f acc[NGATES] = {};
        const int arow = 16 * wave + l15;
        const _Float16* Ah = hthi + (size_t)arow * HDIM;
        const _Float16* Al = htlo + (size_t)arow * HDIM;
        for (int kk = 0; kk < 16; ++kk) {
            const int kb = kk * 32 + halfsel * 8;
            v16h ahi = cat8(*(const v8h*)(Ah + kb), *(const v8h*)(Ah + kb + 16));
            v16h alo = cat8(*(const v8h*)(Al + kb), *(const v8h*)(Al + kb + 16));
#pragma unroll
            for (int g = 0; g < NGATES; ++g) {
                // B-hi from LDS, B-lo from L2: both memory pipes in flight
                const v16h bhi = *(const v16h*)(wvh_s + ((size_t)(g * 16 + kk) * 512 + (size_t)lane * 16));
                const size_t boff = ((size_t)((g * 32 + w) * 16 + kk) * 32 + lane) * 16;
                const v16h blo = *(const v16h*)(WVlo + boff);
                acc[g] = __builtin_amdgcn_wmma_f32_16x16x32_f16(false, ahi, false, bhi, (short)0, acc[g], false, false);
                acc[g] = __builtin_amdgcn_wmma_f32_16x16x32_f16(false, alo, false, bhi, (short)0, acc[g], false, false);
                acc[g] = __builtin_amdgcn_wmma_f32_16x16x32_f16(false, ahi, false, blo, (short)0, acc[g], false, false);
            }
        }
#pragma unroll
        for (int g = 0; g < NGATES; ++g)
#pragma unroll
            for (int v = 0; v < 8; ++v)
                gsm[16 * wave + v + halfsel * 8][g * 16 + l15] = acc[g][v];
        __syncthreads();

        // ---- pointwise CTLSTM update for our 64x16 slice ----
        for (int e = tid; e < BDIM * 16; e += NTHR) {
            const int b = e >> 4, n = e & 15, j = j0 + n;
            const float* wu = WU + (size_t)marks[b * NEV + t] * GCOLS;
            float gv[NGATES];
#pragma unroll
            for (int g = 0; g < NGATES; ++g) {
                const int col = g * HDIM + j;
                gv[g] = gsm[b][g * 16 + n] + wu[col] + bU[col] + bV[col];
            }
            float i_g = sigf(gv[0]), f_g = sigf(gv[1]);
            float ib  = sigf(gv[2]), fb  = sigf(gv[3]);
            float z   = 2.f * sigf(gv[4]);
            float o_g = sigf(gv[5]);
            float dlt = gv[6] > 0.f ? gv[6] : 0.01f * gv[6];
            float ct = ct_s[b][n], cb = cb_s[b][n];
            float clow = f_g * ct + i_g * z;
            float cbn  = fb * cb + ib * z;
            float dt   = times[b * (NEV + 1) + t + 1] - times[b * (NEV + 1) + t];
            float ctn  = cbn + (clow - cbn) * expf(dt * dlt);
            float hn   = o_g * (2.f * sigf(2.f * ctn) - 1.f);
            ct_s[b][n] = ctn;
            cb_s[b][n] = cbn;
            const size_t oidx = ((size_t)b * NEV + t) * HDIM + j;
            out_clow[oidx]  = clow;
            out_cbar[oidx]  = cbn;
            out_delta[oidx] = dlt;
            out_o[oidx]     = o_g;
            _Float16 hh = (_Float16)hn;
            const size_t hidx = (size_t)((t + 1) & 1) * (BDIM * HDIM) + (size_t)b * HDIM + j;
            HTHI[hidx] = hh;
            HTLO[hidx] = (_Float16)(hn - (float)hh);
        }

        // ---- grid barrier: publish ht(t+1) to all WGs ----
        __threadfence();
        __syncthreads();
        if (tid == 0) {
            __hip_atomic_fetch_add(counter, 1u, __ATOMIC_RELEASE, __HIP_MEMORY_SCOPE_AGENT);
            const unsigned target = (unsigned)(t + 1) * NWG;
            while (__hip_atomic_load(counter, __ATOMIC_ACQUIRE, __HIP_MEMORY_SCOPE_AGENT) < target)
                __builtin_amdgcn_s_sleep(2);
        }
        __syncthreads();
    }

    // lambda for the final step (reads ht(NEV) = buffer 0)
    lamb_tile(NEV - 1, r2, c2, wave, lane, tid,
              HTHI + (size_t)(NEV & 1) * (BDIM * HDIM),
              HTLO + (size_t)(NEV & 1) * (BDIM * HDIM),
              WLhi, WLlo, bLam, scale, lpart, out_lamb);
}

// ============================================================================
extern "C" void kernel_launch(void* const* d_in, const int* in_sizes, int n_in,
                              void* d_out, int out_size, void* d_ws, size_t ws_size,
                              hipStream_t stream) {
    const float* seq   = (const float*)d_in[0];
    const float* times = (const float*)d_in[1];
    const float* WU    = (const float*)d_in[2];
    const float* bU    = (const float*)d_in[3];
    const float* WV    = (const float*)d_in[4];
    const float* bV    = (const float*)d_in[5];
    const float* WL    = (const float*)d_in[6];
    const float* bL    = (const float*)d_in[7];
    const float* scale = (const float*)d_in[8];

    char* ws = (char*)d_ws;
    unsigned*  counter = (unsigned*)(ws + OFF_CNT);
    int*       marks   = (int*)(ws + OFF_MARKS);
    _Float16*  WVhi    = (_Float16*)(ws + OFF_WVHI);
    _Float16*  WVlo    = (_Float16*)(ws + OFF_WVLO);
    _Float16*  WLhi    = (_Float16*)(ws + OFF_WLHI);
    _Float16*  WLlo    = (_Float16*)(ws + OFF_WLLO);
    _Float16*  HTHI    = (_Float16*)(ws + OFF_HTHI);
    _Float16*  HTLO    = (_Float16*)(ws + OFF_HTLO);

    float* out       = (float*)d_out;
    float* out_lamb  = out;
    float* out_clow  = out_lamb + (size_t)BDIM * NEV * KDIM;
    float* out_cbar  = out_clow + (size_t)BDIM * NEV * HDIM;
    float* out_delta = out_cbar + (size_t)BDIM * NEV * HDIM;
    float* out_o     = out_delta + (size_t)BDIM * NEV * HDIM;

    ctlstm_init_kernel<<<64, 256, 0, stream>>>(seq, marks, HTHI, HTLO, counter);
    ctlstm_pack_kernel<<<512, 256, 0, stream>>>(WV, WL, WVhi, WVlo, WLhi, WLlo);
    ctlstm_persistent_kernel<<<NWG, NTHR, 0, stream>>>(
        times, WU, bU, bV, bL, scale, marks,
        WVhi, WVlo, WLhi, WLlo, HTHI, HTLO,
        out_lamb, out_clow, out_cbar, out_delta, out_o, counter);
}
